// Decoder_69844758167959
// MI455X (gfx1250) — compile-verified, matrix operand order checked by reading, not run on hardware
//
#include <hip/hip_runtime.h>
#include <cstdint>

// ---------------------------------------------------------------------------
// Sizes (fixed by the reference)
// ---------------------------------------------------------------------------
constexpr int kH   = 1024;   // hidden
constexpr int kMel = 128;    // mel bins
constexpr int kB   = 32;     // batch
constexpr int kTe  = 512;    // encoder time steps
constexpr int kT   = 400;    // decoder steps

typedef __attribute__((ext_vector_type(16))) __bf16 v16bf;
typedef __attribute__((ext_vector_type(8)))  float  v8f;
typedef unsigned short u16;
typedef unsigned int   u32;

// ---------------------------------------------------------------------------
// bf16 helpers (truncation rounding; accumulate in f32)
// ---------------------------------------------------------------------------
__device__ __forceinline__ u32 pack_bf2(float lo, float hi) {
  u32 a = __builtin_bit_cast(u32, lo);
  u32 b = __builtin_bit_cast(u32, hi);
  return (b & 0xFFFF0000u) | (a >> 16);
}
__device__ __forceinline__ u16 f2bf(float v) {
  return (u16)(__builtin_bit_cast(u32, v) >> 16);
}
__device__ __forceinline__ float bflo(u32 u) { return __builtin_bit_cast(float, u << 16); }
__device__ __forceinline__ float bfhi(u32 u) { return __builtin_bit_cast(float, u & 0xFFFF0000u); }
__device__ __forceinline__ float sigmoidf(float x) { return 1.0f / (1.0f + __expf(-x)); }

union AFrag { v16bf v; u32 u[8]; uint4 q[2]; };

// A-fragment (16x32 bf16) straight from a row-major bf16 matrix: zero VALU.
// ISA layout (16-bit A): lanes 0-15 row M=lane, K=0..7 / 16..23; lanes 16-31
// row M=lane-16, K=8..15 / 24..31 -> two contiguous 16B loads per lane.
__device__ __forceinline__ v16bf make_a_bf16(const u16* __restrict__ X, int ld,
                                             int m0, int k0) {
  int lane = threadIdx.x & 31;
  const u16* p = X + (size_t)(m0 + (lane & 15)) * ld + k0 + ((lane & 16) ? 8 : 0);
  AFrag r;
  r.q[0] = *(const uint4*)p;
  r.q[1] = *(const uint4*)(p + 16);
  return r.v;
}

// Pre-packed B tile: 32 lanes x 16 bf16, contiguous 1KB per (k0,n0) tile.
__device__ __forceinline__ v16bf load_b(const u16* __restrict__ tiles, long tileIdx) {
  const v16bf* p = (const v16bf*)tiles;
  return p[tileIdx * 32 + (threadIdx.x & 31)];
}

__device__ __forceinline__ v8f wmma_bf16(v16bf a, v16bf b, v8f c) {
  return __builtin_amdgcn_wmma_f32_16x16x32_bf16(false, a, false, b, (short)0, c, false, false);
}

// ---------------------------------------------------------------------------
// Prep kernels
// ---------------------------------------------------------------------------
// Pack torch-style W [N,K] (f32, row-major) into per-tile WMMA B-register layout
// (bf16). Tile = (k0 step 32, n0 step 16) -> 512 bf16 stored lane-major.
__global__ void pack_w_kernel(const float* __restrict__ W, u16* __restrict__ out,
                              int N, int K) {
  long idx = (long)blockIdx.x * blockDim.x + threadIdx.x;
  long tot = (long)N * K;
  if (idx >= tot) return;
  int n = (int)(idx / K);
  int k = (int)(idx - (long)n * K);
  int tn = n >> 4, nl = n & 15;
  int tk = k >> 5, kk = k & 31;
  int grp  = ((kk & 15) >= 8) ? 1 : 0;
  int lane = nl + (grp << 4);
  int i    = ((kk & 16) ? 8 : 0) + (kk & 7);
  long tileIdx = (long)tn * (K >> 5) + tk;
  out[tileIdx * 512 + lane * 16 + i] = f2bf(W[(long)n * K + k]);
}

__global__ void cvt_bf16_kernel(const float* __restrict__ src, u16* __restrict__ dst, long n) {
  long i = (long)blockIdx.x * blockDim.x + threadIdx.x;
  if (i < n) dst[i] = f2bf(src[i]);
}
__global__ void copy_f32_kernel(const float* __restrict__ s, float* __restrict__ d, int n) {
  int i = blockIdx.x * blockDim.x + threadIdx.x;
  if (i < n) d[i] = s[i];
}
__global__ void zero_u16_kernel(u16* __restrict__ d, int n) {
  int i = blockIdx.x * blockDim.x + threadIdx.x;
  if (i < n) d[i] = 0;
}

// ---------------------------------------------------------------------------
// energy = enc @ W_attn^T + b_attn   -> bf16 [B*Te, H]
// grid (B*Te/32, H/128), 256 threads; wave owns 1 n-tile x 2 m-tiles.
// ---------------------------------------------------------------------------
__global__ void energy_kernel(const u16* __restrict__ encb, const u16* __restrict__ Wt,
                              const float* __restrict__ bias, u16* __restrict__ out) {
  int w = threadIdx.x >> 5, lane = threadIdx.x & 31;
  int m0 = blockIdx.x * 32;
  int n0 = blockIdx.y * 128 + w * 16;
  long tb = (long)(n0 >> 4) * (kH >> 5);
  v8f acc0 = {}, acc1 = {};
#pragma unroll 2
  for (int k0 = 0; k0 < kH; k0 += 32) {
    long ti = tb + (k0 >> 5);
    __builtin_prefetch((const void*)((const v16bf*)Wt + (ti + 1) * 32), 0, 1);
    v16bf bt = load_b(Wt, ti);
    acc0 = wmma_bf16(make_a_bf16(encb, kH, m0, k0), bt, acc0);
    acc1 = wmma_bf16(make_a_bf16(encb, kH, m0 + 16, k0), bt, acc1);
  }
  int n = n0 + (lane & 15);
  float bv = bias[n];
  int mb = m0 + (lane >> 4) * 8;
#pragma unroll
  for (int j = 0; j < 8; ++j) {
    out[(size_t)(mb + j) * kH + n]      = f2bf(acc0[j] + bv);
    out[(size_t)(mb + 16 + j) * kH + n] = f2bf(acc1[j] + bv);
  }
}

// ---------------------------------------------------------------------------
// Fused GRU cell: h' = (1-z)*n + z*h  (torch gate order r,z,n)
// grid H/64, 128 threads (4 waves). Wave owns one n-tile and BOTH m-tiles.
// gi+gh fused into shared accumulators for r,z; inn/hn kept separate.
// Reads bf16 activations (x, h) and packed bf16 B tiles; writes f32+bf16 h'.
// ---------------------------------------------------------------------------
__global__ void gru_kernel(const u16* __restrict__ xb, int Kx,
                           const float* __restrict__ hf, const u16* __restrict__ hb,
                           const u16* __restrict__ Wih, const u16* __restrict__ Whh,
                           const float* __restrict__ bih, const float* __restrict__ bhh,
                           float* __restrict__ hnf, u16* __restrict__ hnb) {
  int w = threadIdx.x >> 5, lane = threadIdx.x & 31;
  int n0 = blockIdx.x * 64 + w * 16;
  v8f aR0 = {}, aZ0 = {}, aI0 = {}, aG0 = {};
  v8f aR1 = {}, aZ1 = {}, aI1 = {}, aG1 = {};

  int tKx = Kx >> 5;
  long ir = (long)((0 * kH + n0) >> 4) * tKx;
  long iz = (long)((1 * kH + n0) >> 4) * tKx;
  long in_ = (long)((2 * kH + n0) >> 4) * tKx;
#pragma unroll 2
  for (int k0 = 0; k0 < Kx; k0 += 32) {
    int kt = k0 >> 5;
    __builtin_prefetch((const void*)((const v16bf*)Wih + (ir + kt + 1) * 32), 0, 1);
    v16bf a0 = make_a_bf16(xb, Kx, 0, k0);
    v16bf a1 = make_a_bf16(xb, Kx, 16, k0);
    v16bf Br = load_b(Wih, ir + kt);
    v16bf Bz = load_b(Wih, iz + kt);
    v16bf Bn = load_b(Wih, in_ + kt);
    aR0 = wmma_bf16(a0, Br, aR0); aR1 = wmma_bf16(a1, Br, aR1);
    aZ0 = wmma_bf16(a0, Bz, aZ0); aZ1 = wmma_bf16(a1, Bz, aZ1);
    aI0 = wmma_bf16(a0, Bn, aI0); aI1 = wmma_bf16(a1, Bn, aI1);
  }

  int tKh = kH >> 5;
  long hr = (long)((0 * kH + n0) >> 4) * tKh;
  long hz = (long)((1 * kH + n0) >> 4) * tKh;
  long hn = (long)((2 * kH + n0) >> 4) * tKh;
#pragma unroll 2
  for (int k0 = 0; k0 < kH; k0 += 32) {
    int kt = k0 >> 5;
    __builtin_prefetch((const void*)((const v16bf*)Whh + (hr + kt + 1) * 32), 0, 1);
    v16bf a0 = make_a_bf16(hb, kH, 0, k0);
    v16bf a1 = make_a_bf16(hb, kH, 16, k0);
    v16bf Br = load_b(Whh, hr + kt);
    v16bf Bz = load_b(Whh, hz + kt);
    v16bf Bn = load_b(Whh, hn + kt);
    aR0 = wmma_bf16(a0, Br, aR0); aR1 = wmma_bf16(a1, Br, aR1);   // r: gi+gh share acc
    aZ0 = wmma_bf16(a0, Bz, aZ0); aZ1 = wmma_bf16(a1, Bz, aZ1);   // z: gi+gh share acc
    aG0 = wmma_bf16(a0, Bn, aG0); aG1 = wmma_bf16(a1, Bn, aG1);   // hn separate
  }

  int n = n0 + (lane & 15);
  float br  = bih[n] + bhh[n];
  float bz  = bih[kH + n] + bhh[kH + n];
  float bin = bih[2 * kH + n];
  float bhn = bhh[2 * kH + n];
  int mb = (lane >> 4) * 8;
#pragma unroll
  for (int j = 0; j < 8; ++j) {
    int m = mb + j;
    float r  = sigmoidf(aR0[j] + br);
    float z  = sigmoidf(aZ0[j] + bz);
    float nn = tanhf(aI0[j] + bin + r * (aG0[j] + bhn));
    float ho = hf[(size_t)m * kH + n];
    float v  = (1.0f - z) * nn + z * ho;
    hnf[(size_t)m * kH + n] = v;
    hnb[(size_t)m * kH + n] = f2bf(v);
  }
#pragma unroll
  for (int j = 0; j < 8; ++j) {
    int m = 16 + mb + j;
    float r  = sigmoidf(aR1[j] + br);
    float z  = sigmoidf(aZ1[j] + bz);
    float nn = tanhf(aI1[j] + bin + r * (aG1[j] + bhn));
    float ho = hf[(size_t)m * kH + n];
    float v  = (1.0f - z) * nn + z * ho;
    hnf[(size_t)m * kH + n] = v;
    hnb[(size_t)m * kH + n] = f2bf(v);
  }
}

// ---------------------------------------------------------------------------
// Attention: scores = energy[b] @ h1[b]; softmax; context = probs @ enc[b]
// grid = B (one workgroup per batch element), 256 threads (8 waves).
// Writes context as packed bf16 (only consumer is the co GEMM A-side).
// ---------------------------------------------------------------------------
__global__ void attn_kernel(const float* __restrict__ h1, const u16* __restrict__ energy,
                            const u16* __restrict__ encbf, u16* __restrict__ ctxb) {
  __shared__ float h1s[kH];
  __shared__ float sc[kTe];
  __shared__ float red[16];
  int b = blockIdx.x, tid = threadIdx.x;
  int w = tid >> 5, lane = tid & 31;

  for (int i = tid; i < kH; i += 256) h1s[i] = h1[(size_t)b * kH + i];
  __syncthreads();

  const u32* Eb = (const u32*)energy + (size_t)b * kTe * (kH / 2);
  for (int t = w; t < kTe; t += 8) {
    const u32* row = Eb + (size_t)t * (kH / 2);
    float s = 0.0f;
#pragma unroll 4
    for (int kc = 0; kc < kH / 2; kc += 32) {
      u32 u = row[kc + lane];
      int k = 2 * (kc + lane);
      s += bflo(u) * h1s[k] + bfhi(u) * h1s[k + 1];
    }
#pragma unroll
    for (int m = 16; m; m >>= 1) s += __shfl_xor(s, m, 32);
    if (lane == 0) sc[t] = s;
  }
  __syncthreads();

  float mx = -1e30f;
  for (int t = tid; t < kTe; t += 256) mx = fmaxf(mx, sc[t]);
#pragma unroll
  for (int m = 16; m; m >>= 1) mx = fmaxf(mx, __shfl_xor(mx, m, 32));
  if (lane == 0) red[w] = mx;
  __syncthreads();
  float gmx = red[0];
#pragma unroll
  for (int i = 1; i < 8; ++i) gmx = fmaxf(gmx, red[i]);

  float psum = 0.0f;
  for (int t = tid; t < kTe; t += 256) {
    float e = __expf(sc[t] - gmx);
    sc[t] = e;
    psum += e;
  }
#pragma unroll
  for (int m = 16; m; m >>= 1) psum += __shfl_xor(psum, m, 32);
  if (lane == 0) red[8 + w] = psum;
  __syncthreads();
  float tot = 0.0f;
#pragma unroll
  for (int i = 0; i < 8; ++i) tot += red[8 + i];
  float inv = 1.0f / tot;

  int hb = tid * 4;
  float a0 = 0, a1 = 0, a2 = 0, a3 = 0;
  const u32* Cb = (const u32*)encbf + (size_t)b * kTe * (kH / 2) + (hb >> 1);
#pragma unroll 4
  for (int t = 0; t < kTe; ++t) {
    float p = sc[t] * inv;
    const u32* rr = Cb + (size_t)t * (kH / 2);
    u32 u0 = rr[0], u1 = rr[1];
    a0 += p * bflo(u0); a1 += p * bfhi(u0);
    a2 += p * bflo(u1); a3 += p * bfhi(u1);
  }
  u32* cp = (u32*)ctxb + ((size_t)b * kH + hb) / 2;
  cp[0] = pack_bf2(a0, a1);
  cp[1] = pack_bf2(a2, a3);
}

// ---------------------------------------------------------------------------
// co = tanh([h1 | ctx] @ Wc^T + bc)   grid H/64, 128 threads (4 waves)
// Wave owns 1 n-tile x 2 m-tiles; A from bf16 h1 / bf16 ctx. Writes bf16 co.
// ---------------------------------------------------------------------------
__global__ void co_kernel(const u16* __restrict__ h1b, const u16* __restrict__ ctxb,
                          const u16* __restrict__ Wct, const float* __restrict__ bc,
                          u16* __restrict__ cob) {
  int w = threadIdx.x >> 5, lane = threadIdx.x & 31;
  int n0 = blockIdx.x * 64 + w * 16;
  long tb = (long)(n0 >> 4) * ((2 * kH) >> 5);
  v8f acc0 = {}, acc1 = {};
#pragma unroll 2
  for (int k0 = 0; k0 < 2 * kH; k0 += 32) {
    const u16* src = (k0 < kH) ? h1b : ctxb;
    int kk = (k0 < kH) ? k0 : (k0 - kH);
    long ti = tb + (k0 >> 5);
    __builtin_prefetch((const void*)((const v16bf*)Wct + (ti + 1) * 32), 0, 1);
    v16bf bt = load_b(Wct, ti);
    acc0 = wmma_bf16(make_a_bf16(src, kH, 0, kk), bt, acc0);
    acc1 = wmma_bf16(make_a_bf16(src, kH, 16, kk), bt, acc1);
  }
  int n = n0 + (lane & 15);
  float bv = bc[n];
  int mb = (lane >> 4) * 8;
#pragma unroll
  for (int j = 0; j < 8; ++j) {
    cob[(size_t)(mb + j) * kH + n]      = f2bf(tanhf(acc0[j] + bv));
    cob[(size_t)(mb + 16 + j) * kH + n] = f2bf(tanhf(acc1[j] + bv));
  }
}

// ---------------------------------------------------------------------------
// mel = co @ Wp^T + bp  (WMMA, block 0: 8 waves x 1 n-tile x 2 m-tiles)
// gate/mask (block 1, VALU over bf16 co). Writes masked outputs + next decin.
// ---------------------------------------------------------------------------
__global__ void melgate_kernel(const u16* __restrict__ cob, const u16* __restrict__ Wpt,
                               const float* __restrict__ bp, const float* __restrict__ Wg,
                               const float* __restrict__ bg, const int* __restrict__ lens,
                               int t, u16* __restrict__ decinb,
                               float* __restrict__ mels, float* __restrict__ gates,
                               float* __restrict__ maskp) {
  if (blockIdx.x == 0) {
    int w = threadIdx.x >> 5, lane = threadIdx.x & 31;
    int n0 = w * 16;
    long tb = (long)(n0 >> 4) * (kH >> 5);
    v8f acc0 = {}, acc1 = {};
#pragma unroll 2
    for (int k0 = 0; k0 < kH; k0 += 32) {
      long ti = tb + (k0 >> 5);
      __builtin_prefetch((const void*)((const v16bf*)Wpt + (ti + 1) * 32), 0, 1);
      v16bf bt = load_b(Wpt, ti);
      acc0 = wmma_bf16(make_a_bf16(cob, kH, 0, k0), bt, acc0);
      acc1 = wmma_bf16(make_a_bf16(cob, kH, 16, k0), bt, acc1);
    }
    int n = n0 + (lane & 15);
    float bv = bp[n];
    int mb = (lane >> 4) * 8;
#pragma unroll
    for (int j = 0; j < 8; ++j) {
      int m = mb + j;
      float v = acc0[j] + bv;
      decinb[(size_t)m * kMel + n] = f2bf(v);
      bool msk = (t > lens[m]);
      mels[(size_t)m * kT * kMel + (size_t)t * kMel + n] = msk ? 0.0f : v;
    }
#pragma unroll
    for (int j = 0; j < 8; ++j) {
      int m = 16 + mb + j;
      float v = acc1[j] + bv;
      decinb[(size_t)m * kMel + n] = f2bf(v);
      bool msk = (t > lens[m]);
      mels[(size_t)m * kT * kMel + (size_t)t * kMel + n] = msk ? 0.0f : v;
    }
  } else {
    __shared__ float gg[256];
    int tid = threadIdx.x;
    int b = tid >> 3, part = tid & 7;
    float s = 0.0f;
    const u32* cb = (const u32*)cob + (size_t)b * (kH / 2) + part * 64;
#pragma unroll 4
    for (int j = 0; j < 64; ++j) {
      u32 u = cb[j];
      s += bflo(u) * Wg[part * 128 + 2 * j] + bfhi(u) * Wg[part * 128 + 2 * j + 1];
    }
    gg[tid] = s;
    __syncthreads();
    if (part == 0) {
      float tot = 0.0f;
#pragma unroll
      for (int p = 0; p < 8; ++p) tot += gg[b * 8 + p];
      tot += bg[0];
      bool msk = (t > lens[b]);
      gates[(size_t)b * kT + t] = msk ? 1000.0f : tot;
      maskp[(size_t)b * kT + t] = msk ? 1.0f : 0.0f;
    }
  }
}

// ---------------------------------------------------------------------------
// Host driver
// ---------------------------------------------------------------------------
extern "C" void kernel_launch(void* const* d_in, const int* in_sizes, int n_in,
                              void* d_out, int out_size, void* d_ws, size_t ws_size,
                              hipStream_t stream) {
  (void)in_sizes; (void)n_in; (void)out_size; (void)ws_size;
  const float* enc_hidden = (const float*)d_in[0];   // [4,B,H]
  const float* enc_out    = (const float*)d_in[1];   // [B,Te,H]
  const int*   lens       = (const int*)d_in[3];     // [B]
  const float* W_attn = (const float*)d_in[4];
  const float* b_attn = (const float*)d_in[5];
  const float* W_ih0  = (const float*)d_in[6];
  const float* W_hh0  = (const float*)d_in[7];
  const float* b_ih0  = (const float*)d_in[8];
  const float* b_hh0  = (const float*)d_in[9];
  const float* W_ih1  = (const float*)d_in[10];
  const float* W_hh1  = (const float*)d_in[11];
  const float* b_ih1  = (const float*)d_in[12];
  const float* b_hh1  = (const float*)d_in[13];
  const float* Wc     = (const float*)d_in[14];
  const float* bc     = (const float*)d_in[15];
  const float* Wp     = (const float*)d_in[16];
  const float* bp     = (const float*)d_in[17];
  const float* Wg     = (const float*)d_in[18];
  const float* bg     = (const float*)d_in[19];

  char* ws = (char*)d_ws;
  size_t off = 0;
  auto take = [&](size_t bytes) -> char* {
    char* p = ws + off;
    off = (off + bytes + 255) & ~(size_t)255;
    return p;
  };
  u16* Wih0_t  = (u16*)take((size_t)3 * kH * kMel * 2);
  u16* Whh0_t  = (u16*)take((size_t)3 * kH * kH * 2);
  u16* Wih1_t  = (u16*)take((size_t)3 * kH * kH * 2);
  u16* Whh1_t  = (u16*)take((size_t)3 * kH * kH * 2);
  u16* Wattn_t = (u16*)take((size_t)kH * kH * 2);
  u16* Wc_t    = (u16*)take((size_t)kH * 2 * kH * 2);
  u16* Wp_t    = (u16*)take((size_t)kMel * kH * 2);
  u16* energy  = (u16*)take((size_t)kB * kTe * kH * 2);
  u16* encbf   = (u16*)take((size_t)kB * kTe * kH * 2);
  float* h0f   = (float*)take((size_t)2 * kB * kH * 4);
  float* h1f   = (float*)take((size_t)2 * kB * kH * 4);
  u16* h0b     = (u16*)take((size_t)2 * kB * kH * 2);
  u16* h1b     = (u16*)take((size_t)2 * kB * kH * 2);
  u16* decinb  = (u16*)take((size_t)kB * kMel * 2);
  u16* ctxb    = (u16*)take((size_t)kB * kH * 2);
  u16* cob     = (u16*)take((size_t)kB * kH * 2);

  float* mels  = (float*)d_out;
  float* gates = mels + (size_t)kB * kT * kMel;
  float* maskp = gates + (size_t)kB * kT;

  auto pack = [&](const float* W, u16* out, int N, int K) {
    long tot = (long)N * K;
    pack_w_kernel<<<(unsigned)((tot + 255) / 256), 256, 0, stream>>>(W, out, N, K);
  };
  pack(W_ih0, Wih0_t, 3 * kH, kMel);
  pack(W_hh0, Whh0_t, 3 * kH, kH);
  pack(W_ih1, Wih1_t, 3 * kH, kH);
  pack(W_hh1, Whh1_t, 3 * kH, kH);
  pack(W_attn, Wattn_t, kH, kH);
  pack(Wc, Wc_t, kH, 2 * kH);
  pack(Wp, Wp_t, kMel, kH);

  long ne = (long)kB * kTe * kH;
  cvt_bf16_kernel<<<(unsigned)((ne + 255) / 256), 256, 0, stream>>>(enc_out, encbf, ne);
  // initial decoder states: f32 + bf16 copies of encoder_hidden[0] / [1]
  copy_f32_kernel<<<(kB * kH + 255) / 256, 256, 0, stream>>>(enc_hidden, h0f, kB * kH);
  copy_f32_kernel<<<(kB * kH + 255) / 256, 256, 0, stream>>>(enc_hidden + (size_t)kB * kH,
                                                             h1f, kB * kH);
  cvt_bf16_kernel<<<(kB * kH + 255) / 256, 256, 0, stream>>>(enc_hidden, h0b, kB * kH);
  cvt_bf16_kernel<<<(kB * kH + 255) / 256, 256, 0, stream>>>(enc_hidden + (size_t)kB * kH,
                                                             h1b, kB * kH);
  zero_u16_kernel<<<(kB * kMel + 255) / 256, 256, 0, stream>>>(decinb, kB * kMel);

  energy_kernel<<<dim3(kB * kTe / 32, kH / 128), 256, 0, stream>>>(encbf, Wattn_t, b_attn,
                                                                   energy);

  for (int t = 0; t < kT; ++t) {
    int cur = t & 1, nxt = cur ^ 1;
    float* h0fc = h0f + (size_t)cur * kB * kH;
    float* h0fn = h0f + (size_t)nxt * kB * kH;
    float* h1fc = h1f + (size_t)cur * kB * kH;
    float* h1fn = h1f + (size_t)nxt * kB * kH;
    u16* h0bc = h0b + (size_t)cur * kB * kH;
    u16* h0bn = h0b + (size_t)nxt * kB * kH;
    u16* h1bc = h1b + (size_t)cur * kB * kH;
    u16* h1bn = h1b + (size_t)nxt * kB * kH;

    gru_kernel<<<kH / 64, 128, 0, stream>>>(decinb, kMel, h0fc, h0bc, Wih0_t, Whh0_t,
                                            b_ih0, b_hh0, h0fn, h0bn);
    gru_kernel<<<kH / 64, 128, 0, stream>>>(h0bn, kH, h1fc, h1bc, Wih1_t, Whh1_t,
                                            b_ih1, b_hh1, h1fn, h1bn);
    attn_kernel<<<kB, 256, 0, stream>>>(h1fn, energy, encbf, ctxb);
    co_kernel<<<kH / 64, 128, 0, stream>>>(h1bn, ctxb, Wc_t, bc, cob);
    melgate_kernel<<<2, 256, 0, stream>>>(cob, Wp_t, bp, Wg, bg, lens, t, decinb,
                                          mels, gates, maskp);
  }
}
